// MultiHeadAttentionLayer_18726057411021
// MI455X (gfx1250) — matrix operand order repaired and checked by reading, not compiled, not run on hardware
//
#include <hip/hip_runtime.h>

// ---------------------------------------------------------------------------
// CDNA5 (gfx1250) WMMA types
// ---------------------------------------------------------------------------
typedef __attribute__((ext_vector_type(16))) __bf16 v16bf;
typedef __attribute__((ext_vector_type(8)))  float  v8f;

__device__ __forceinline__ unsigned short f32_to_bf16_rne(float f) {
  unsigned int u = __float_as_uint(f);
  u += 0x7FFFu + ((u >> 16) & 1u);        // round-to-nearest-even
  return (unsigned short)(u >> 16);
}
__device__ __forceinline__ float bf16_to_f32(unsigned short h) {
  return __uint_as_float(((unsigned int)h) << 16);
}
__device__ __forceinline__ unsigned int pack_bf16x2(float a, float b) {
  return (unsigned int)f32_to_bf16_rne(a) | ((unsigned int)f32_to_bf16_rne(b) << 16);
}

// ---------------------------------------------------------------------------
// CDNA5 async global->LDS copy (ASYNCcnt-tracked, no VGPR round trip).
// VDST operand is the LDS byte address: for a flat pointer into __shared__,
// addr[31:0] IS the LDS offset (ISA 10.2 aperture rules), so truncation works.
// ---------------------------------------------------------------------------
__device__ __forceinline__ void async_lds_b128(unsigned lds_addr, const void* gaddr) {
  asm volatile("global_load_async_to_lds_b128 %0, %1, off"
               :: "v"(lds_addr), "v"(gaddr) : "memory");
}
__device__ __forceinline__ void wait_async0() {
  asm volatile("s_wait_asynccnt 0x0" ::: "memory");
}

// ---------------------------------------------------------------------------
// Async panel staging: 128 rows x 64 K of bf16, 256 threads x 4 b128 chunks.
// LDS row stride 72 bf16 (144 B): keeps b128 alignment, staggers banks.
// ---------------------------------------------------------------------------
__device__ __forceinline__ void tile_issue_async(unsigned short (*dst)[72],
                                                 const unsigned short* __restrict__ g,
                                                 int ld, int tid) {
#pragma unroll
  for (int i = 0; i < 4; ++i) {
    int c   = tid + i * 256;              // 1024 b128 chunks
    int row = c >> 3;
    int col = (c & 7) << 3;
    async_lds_b128((unsigned)(unsigned long long)&dst[row][col],
                   (const void*)(g + (long long)row * ld + col));
  }
}

// ---------------------------------------------------------------------------
// Tiled bf16 WMMA GEMM:  Out[b,m,n] = alpha * sum_k A[b,m,k]*B[b,n,k] (+bias[n])
// Block tile 128x128, 8 waves (wave32) in a 2(M) x 4(N) grid, 64x32 per wave.
// v_wmma_f32_16x16x32_bf16 with fp32 accumulation. Both panels stream with
// global_load_async_to_lds_b128 into double-buffered LDS (2 x 64-K, 74 KB),
// overlapping global traffic with the 16 WMMAs per panel. All operands bf16,
// so no register staging -> no VGPR-MSB pressure.
// All problem dims are multiples of the tile sizes (4096 / 512), no bounds.
// ---------------------------------------------------------------------------
template<bool OUT_BF16, bool HAS_BIAS>
__global__ __launch_bounds__(256)
void gemm_wmma_bf16(const unsigned short* __restrict__ A,
                    const unsigned short* __restrict__ B,
                    const float* __restrict__ bias, void* __restrict__ Outp,
                    int K, int lda, int ldb, int ldo,
                    long long aBatch, long long bBatch, long long oBatch,
                    float alpha) {
  constexpr int KT = 64;
  __shared__ unsigned short lA[2][128][72];
  __shared__ unsigned short lB[2][128][72];

  const int tid   = threadIdx.x;
  const int lane  = tid & 31;
  const int wave  = tid >> 5;
  const int waveM = (wave & 1) * 64;    // 2 waves along M
  const int waveN = (wave >> 1) * 32;   // 4 waves along N
  const int lmod  = lane & 15;
  const int halfS = lane >> 4;          // lane half-select (ISA frag layout)

  const long long blockM = (long long)blockIdx.y * 128;
  const long long blockN = (long long)blockIdx.x * 128;
  const long long zb     = blockIdx.z;
  const unsigned short* gA = A + zb * aBatch + blockM * (long long)lda;
  const unsigned short* gB = B + zb * bBatch + blockN * (long long)ldb;

  union AFrag { uint4 u[2]; v16bf v; };
  union BFrag { uint4 u[2]; v16bf v; };
  union Acc   { v8f v; float f[8]; };

  Acc acc[4][2];
#pragma unroll
  for (int mt = 0; mt < 4; ++mt)
#pragma unroll
    for (int nt = 0; nt < 2; ++nt)
#pragma unroll
      for (int r = 0; r < 8; ++r) acc[mt][nt].f[r] = 0.0f;

  const int nk = K / KT;

  // Prologue: panel 0 -> buffer 0
  tile_issue_async(lA[0], gA, lda, tid);
  tile_issue_async(lB[0], gB, ldb, tid);
  wait_async0();
  __syncthreads();

  for (int p = 0; p < nk; ++p) {
    const int buf = p & 1;
    unsigned short (*cA)[72] = lA[buf];
    unsigned short (*cB)[72] = lB[buf];

    // Kick off panel p+1 into the other buffer (safe: every wave passed the
    // barrier ending iteration p-1, which retired all reads of that buffer).
    if (p + 1 < nk) {
      tile_issue_async(lA[buf ^ 1], gA + (p + 1) * KT, lda, tid);
      tile_issue_async(lB[buf ^ 1], gB + (p + 1) * KT, ldb, tid);
    }

#pragma unroll
    for (int ks = 0; ks < KT; ks += 32) {
      AFrag a[4];
      BFrag b[2];
      // A 16x32 bf16 frag: VGPR 0..3 <- K = ks+half*8+{0..7};
      //                    VGPR 4..7 <- K = ks+16+half*8+{0..7}
#pragma unroll
      for (int mt = 0; mt < 4; ++mt) {
        const int m = waveM + mt * 16 + lmod;
        a[mt].u[0] = *(const uint4*)&cA[m][ks + halfS * 8];
        a[mt].u[1] = *(const uint4*)&cA[m][ks + 16 + halfS * 8];
      }
      // B 32x16 bf16 frag: VGPR 0..3 <- K = ks+half*8+{0..7}; VGPR 4..7 <- K = ks+16+half*8+{0..7}
#pragma unroll
      for (int nt = 0; nt < 2; ++nt) {
        const int n = waveN + nt * 16 + lmod;
        b[nt].u[0] = *(const uint4*)&cB[n][ks + halfS * 8];
        b[nt].u[1] = *(const uint4*)&cB[n][ks + 16 + halfS * 8];
      }
#pragma unroll
      for (int mt = 0; mt < 4; ++mt)
#pragma unroll
        for (int nt = 0; nt < 2; ++nt)
          acc[mt][nt].v = __builtin_amdgcn_wmma_f32_16x16x32_bf16(
              false, a[mt].v, false, b[nt].v, (short)0, acc[mt][nt].v,
              false, false);
    }

    if (p + 1 < nk) {
      wait_async0();      // this wave's panel p+1 ops landed in LDS
      __syncthreads();    // everyone's panel p+1 visible; panel p reads retired
    }
  }

  // C/D layout: lane n=lane&15, rows m = (lane>>4)*8 + vgpr
  const long long outBase = zb * oBatch;
#pragma unroll
  for (int nt = 0; nt < 2; ++nt) {
    const long long col = blockN + waveN + nt * 16 + lmod;
    const float bv = HAS_BIAS ? bias[col] : 0.0f;
#pragma unroll
    for (int mt = 0; mt < 4; ++mt) {
#pragma unroll
      for (int r = 0; r < 8; ++r) {
        const long long row = blockM + waveM + mt * 16 + halfS * 8 + r;
        const float val = acc[mt][nt].f[r] * alpha + bv;
        if constexpr (OUT_BF16)
          ((unsigned short*)Outp)[outBase + row * (long long)ldo + col] =
              f32_to_bf16_rne(val);
        else
          ((float*)Outp)[outBase + row * (long long)ldo + col] = val;
      }
    }
  }
}

// ---------------------------------------------------------------------------
// fp32 -> bf16 cast, 8 elems/thread (two float4 in, one uint4 out).
// Purely bandwidth bound; keeps the GEMM free of conversion register traffic.
// ---------------------------------------------------------------------------
__global__ __launch_bounds__(256)
void cast_f32_bf16(const float* __restrict__ src, unsigned short* __restrict__ dst) {
  const long long i = ((long long)blockIdx.x * 256 + threadIdx.x) * 8;
  const float4 a = *(const float4*)(src + i);
  const float4 b = *(const float4*)(src + i + 4);
  uint4 o;
  o.x = pack_bf16x2(a.x, a.y);
  o.y = pack_bf16x2(a.z, a.w);
  o.z = pack_bf16x2(b.x, b.y);
  o.w = pack_bf16x2(b.z, b.w);
  *(uint4*)(dst + i) = o;
}

// ---------------------------------------------------------------------------
// Softmax over axis=2 (the S axis) of scores (NH batches of 512x512 fp32):
// per column t, softmax over the 512 rows. Lane index = column -> every row
// sweep is fully coalesced.
// ---------------------------------------------------------------------------
__global__ __launch_bounds__(256)
void softmax_over_s(float* __restrict__ scores) {
  const int t = blockIdx.x * 256 + threadIdx.x;              // 0..511
  float* p = scores + (long long)blockIdx.y * (512ll * 512) + t;
  float m = -3.4e38f;
  for (int s = 0; s < 512; ++s) m = fmaxf(m, p[(long long)s * 512]);
  float sum = 0.0f;
  for (int s = 0; s < 512; ++s) sum += __expf(p[(long long)s * 512] - m);
  const float inv = 1.0f / sum;
  for (int s = 0; s < 512; ++s) {
    const long long o = (long long)s * 512;
    p[o] = __expf(p[o] - m) * inv;
  }
}

// ---------------------------------------------------------------------------
// y = attn * v (elementwise, shapes coincide since S==T==HD), then the
// (0,2,1,3) transpose + reshape to (N, T, D), emitted as bf16 for the final
// WMMA GEMM. attn/v flat layout: ((n*H+h)*512 + i)*512 + j.
// Output: n*2^21 + i*4096 + h*512 + j.
// ---------------------------------------------------------------------------
__global__ __launch_bounds__(256)
void mul_transpose_bf16(const float* __restrict__ attn,
                        const unsigned short* __restrict__ vb,
                        unsigned short* __restrict__ yb) {
  const long long idx = (long long)blockIdx.x * 256 + threadIdx.x;
  const int j = (int)(idx & 511);
  const int i = (int)((idx >> 9) & 511);
  const int h = (int)((idx >> 18) & 7);
  const long long n = idx >> 21;
  const float y = attn[idx] * bf16_to_f32(vb[idx]);
  yb[(n << 21) + ((long long)i << 12) + (h << 9) + j] = f32_to_bf16_rne(y);
}

// ---------------------------------------------------------------------------
// Pipeline (all on `stream`, stream order = dependency order):
//   per projection X in {q,k,v}: cast(x)->xb, cast(W)->wb, GEMM -> {q,k,v}b
//   scores = qb @ kb^T / sqrt(512) (64 batched 512^3 GEMMs, contiguous slices)
//   column softmax (axis=2), y = attn*v + (0,2,1,3) reshape -> yb (alias xb)
//   cast(Wo)->wb, out = yb @ wb^T + bo (fp32)
// Workspace: xb(32M) wb(32M) qb/kb/vb(96M) scores(64M) = 224 MB.
// ---------------------------------------------------------------------------
extern "C" void kernel_launch(void* const* d_in, const int* in_sizes, int n_in,
                              void* d_out, int out_size, void* d_ws,
                              size_t ws_size, hipStream_t stream) {
  const float* query = (const float*)d_in[0];
  const float* key_  = (const float*)d_in[1];
  const float* value = (const float*)d_in[2];
  const float* Wq    = (const float*)d_in[3];
  const float* bq    = (const float*)d_in[4];
  const float* Wk    = (const float*)d_in[5];
  const float* bk    = (const float*)d_in[6];
  const float* Wv    = (const float*)d_in[7];
  const float* bv    = (const float*)d_in[8];
  const float* Wo    = (const float*)d_in[9];
  const float* bo    = (const float*)d_in[10];
  float* out = (float*)d_out;

  char* ws = (char*)d_ws;
  const long long SZ = 33554432ll;                 // 16.7M bf16 = 32 MB
  unsigned short* xb = (unsigned short*)(ws);          // casted activation
  unsigned short* wb = (unsigned short*)(ws + SZ);     // casted weight
  unsigned short* qb = (unsigned short*)(ws + 2 * SZ);
  unsigned short* kb = (unsigned short*)(ws + 3 * SZ);
  unsigned short* vb = (unsigned short*)(ws + 4 * SZ);
  float* scores      = (float*)(ws + 5 * SZ);          // 64 MB fp32
  unsigned short* yb = xb;                             // alias: xb dead by then

  const dim3 blk(256);
  const dim3 gCast(8192);                          // 16.7M elems / (256*8)
  const dim3 gProj(32, 32, 1);                     // 4096/128 x 4096/128

  // Projections: out = X @ W^T + b  (cast to bf16, then async-WMMA GEMM)
  cast_f32_bf16<<<gCast, blk, 0, stream>>>(query, xb);
  cast_f32_bf16<<<gCast, blk, 0, stream>>>(Wq, wb);
  gemm_wmma_bf16<true, true><<<gProj, blk, 0, stream>>>(
      xb, wb, bq, qb, 4096, 4096, 4096, 4096, 0, 0, 0, 1.0f);

  cast_f32_bf16<<<gCast, blk, 0, stream>>>(key_, xb);
  cast_f32_bf16<<<gCast, blk, 0, stream>>>(Wk, wb);
  gemm_wmma_bf16<true, true><<<gProj, blk, 0, stream>>>(
      xb, wb, bk, kb, 4096, 4096, 4096, 4096, 0, 0, 0, 1.0f);

  cast_f32_bf16<<<gCast, blk, 0, stream>>>(value, xb);
  cast_f32_bf16<<<gCast, blk, 0, stream>>>(Wv, wb);
  gemm_wmma_bf16<true, true><<<gProj, blk, 0, stream>>>(
      xb, wb, bv, vb, 4096, 4096, 4096, 4096, 0, 0, 0, 1.0f);

  // scores[nh] = (q2 @ k2^T) / sqrt(512); each (n,h) slice is a contiguous
  // 512x512 matrix (the .view is a contiguous reshape, not a head transpose).
  const float inv_sqrt_hd = 0.04419417382415922f;  // 1/sqrt(512)
  const dim3 gScore(4, 4, 64);
  gemm_wmma_bf16<false, false><<<gScore, blk, 0, stream>>>(
      qb, kb, nullptr, scores, 512, 512, 512, 512,
      262144ll, 262144ll, 262144ll, inv_sqrt_hd);

  // softmax over the S axis (rows within each column), in place
  softmax_over_s<<<dim3(2, 64), blk, 0, stream>>>(scores);

  // y = attn * v, then (0,2,1,3)-transpose/reshape into bf16 yb
  mul_transpose_bf16<<<dim3(65536), blk, 0, stream>>>(scores, vb, yb);

  // out = y @ Wo^T + bo (fp32 output)
  cast_f32_bf16<<<gCast, blk, 0, stream>>>(Wo, wb);
  gemm_wmma_bf16<false, true><<<gProj, blk, 0, stream>>>(
      yb, wb, bo, out, 4096, 4096, 4096, 4096, 0, 0, 0, 1.0f);
}
